// StackedContextViT_70033736729170
// MI455X (gfx1250) — matrix operand order repaired and checked by reading, not compiled, not run on hardware
//
#include <hip/hip_runtime.h>
#include <hip/hip_bf16.h>

// ---------------------------------------------------------------------------
// StackedContextViT forward for gfx1250 (MI455X).
// GEMMs + attention use v_wmma_f32_16x16x32_bf16 (wave32 WMMA).
// GEMM LDS staging uses global_load_async_to_lds_b128 + s_wait_asynccnt
// with double buffering (CDNA5 async memory path).
// ---------------------------------------------------------------------------

typedef __attribute__((ext_vector_type(16))) __bf16 v16bf;
typedef __attribute__((ext_vector_type(8)))  float  v8f;

union Frag16 { v16bf v; uint4 q[2]; };            // 16 bf16 operand fragment
union AccU   { v8f v; float f[8]; };              // 16x16 f32 C/D fragment

__device__ __forceinline__ unsigned short f2bf(float x) {
  unsigned int u = __float_as_uint(x);
  unsigned int r = (u + 0x7FFFu + ((u >> 16) & 1u)) >> 16;   // RNE
  return (unsigned short)r;
}

__device__ __forceinline__ v8f wmma_bf16(const v16bf& a, const v16bf& b, v8f c) {
  // (neg_a, A, neg_b, B, c_mod, C, reuse_a, reuse_b)
  return __builtin_amdgcn_wmma_f32_16x16x32_bf16(false, a, false, b, (short)0, c,
                                                 false, false);
}

// Async DMA one 16-byte chunk: global -> LDS (ASYNCcnt-tracked).
// lds_off: workgroup-relative LDS byte offset (low 32 bits of generic ptr).
__device__ __forceinline__ void async_ld16(unsigned lds_off, const void* g) {
  asm volatile("global_load_async_to_lds_b128 %0, %1, off"
               :: "v"(lds_off), "v"(g) : "memory");
}

// ------------------------------ elementwise --------------------------------

__global__ void k_f32_to_bf16(const float* __restrict__ s,
                              unsigned short* __restrict__ d, size_t n) {
  for (size_t i = (size_t)blockIdx.x * blockDim.x + threadIdx.x; i < n;
       i += (size_t)gridDim.x * blockDim.x)
    d[i] = f2bf(s[i]);
}

__global__ void k_patch_prep(const float* __restrict__ patches,
                             const int* __restrict__ mask,
                             const float* __restrict__ mtok,
                             unsigned short* __restrict__ pin, int total) {
  for (int i = blockIdx.x * blockDim.x + threadIdx.x; i < total;
       i += gridDim.x * blockDim.x) {
    int p   = i % 768;
    int tok = i / 768;                    // b*N + n
    float v = mask[tok] ? mtok[p] : patches[i];
    pin[i] = f2bf(v);
  }
}

__global__ void k_rel_bias(const float* __restrict__ rel, float* __restrict__ bNN) {
  int i = blockIdx.x * blockDim.x + threadIdx.x;
  if (i >= 1024 * 1024) return;
  int q = i >> 10, k = i & 1023;
  int rq = q >> 5, cq = q & 31, rk = k >> 5, ck = k & 31;
  int ir = rq - rk + 31, ic = cq - ck + 31;
  bNN[i] = rel[ir * 63 + ic];
}

// vT[b][h][d][n] = V from qkv (bf16), so attention B-frags are key-contiguous
__global__ void k_build_vt(const unsigned short* __restrict__ qkv,
                           unsigned short* __restrict__ vT, int total) {
  for (int i = blockIdx.x * blockDim.x + threadIdx.x; i < total;
       i += gridDim.x * blockDim.x) {
    int n = i & 1023;
    int d = (i >> 10) & 63;
    int h = (i >> 16) % 12;
    int b = i / (1024 * 64 * 12);
    vT[i] = qkv[((size_t)(b * 1024 + n)) * 2304 + 1536 + h * 64 + d];
  }
}

__global__ void k_mask_out(const int* __restrict__ mask, float* __restrict__ o, int n) {
  int i = blockIdx.x * blockDim.x + threadIdx.x;
  if (i < n) o[i] = mask[i] ? 1.0f : 0.0f;
}

// ------------------------------ GEMM (WMMA) --------------------------------
// C[M,Nout] = A[M,K](bf16) * W[Nout,K]^T(bf16) + bias, optional pos-add/GELU.
// Block tile 128x64, 8 waves (4x2), each wave 32x32 (2x2 WMMA frags), K-step 32.
// Double-buffered LDS filled by async global->LDS DMA.

#define TM 128
#define TN 64
#define TK 32
#define GF_GELU 1
#define GF_POS  2

__global__ __launch_bounds__(256) void k_gemm_bf16(
    const unsigned short* __restrict__ A, const unsigned short* __restrict__ W,
    const float* __restrict__ bias, float* __restrict__ outF,
    unsigned short* __restrict__ outB, const float* __restrict__ pos,
    int M, int K, int Nout, int flags) {
  __shared__ __align__(16) unsigned short As[2][TM * TK];
  __shared__ __align__(16) unsigned short Bs[2][TN * TK];
  const int tid  = threadIdx.x;
  const int lane = tid & 31, wid = tid >> 5;
  const int wm = wid & 3, wn = wid >> 2;        // 4 waves in M, 2 in N
  const int mBase = blockIdx.y * TM, nBase = blockIdx.x * TN;
  const int l15 = lane & 15, hi = lane >> 4;

  // per-thread staging coordinates (fixed across k-steps)
  const int ar0 = tid >> 2,         acc0 = (tid & 3) * 8;
  const int ar1 = (tid + 256) >> 2, acc1 = ((tid + 256) & 3) * 8;
  const int br  = tid >> 2,         bcc  = (tid & 3) * 8;

  auto stage = [&](int buf, int kb) {
    async_ld16((unsigned)(size_t)&As[buf][ar0 * TK + acc0],
               &A[(size_t)(mBase + ar0) * K + kb + acc0]);
    async_ld16((unsigned)(size_t)&As[buf][ar1 * TK + acc1],
               &A[(size_t)(mBase + ar1) * K + kb + acc1]);
    async_ld16((unsigned)(size_t)&Bs[buf][br * TK + bcc],
               &W[(size_t)(nBase + br) * K + kb + bcc]);
  };

  AccU acc[2][2];
  for (int i = 0; i < 2; ++i)
    for (int j = 0; j < 2; ++j)
      for (int r = 0; r < 8; ++r) acc[i][j].f[r] = 0.f;

  stage(0, 0);
  int cur = 0;
  for (int kb = 0; kb < K; kb += TK) {
    const int nk = kb + TK;
    if (nk < K) {
      stage(cur ^ 1, nk);                       // prefetch next tile (async)
      asm volatile("s_wait_asynccnt 0x3" ::: "memory");  // current tile done
    } else {
      asm volatile("s_wait_asynccnt 0x0" ::: "memory");
    }
    __syncthreads();                            // all waves' DMA visible

    // A fragment: lane l15 = row, klo in {0,8}; chunks at klo and klo+16
    Frag16 a[2], b2[2];
    const int aklo = hi ? 8 : 0;
    for (int i = 0; i < 2; ++i) {
      int row = wm * 32 + i * 16 + l15;
      a[i].q[0] = *(const uint4*)&As[cur][row * TK + aklo];
      a[i].q[1] = *(const uint4*)&As[cur][row * TK + aklo + 16];
    }
    // B fragment: lane l15 = out-col, klo in {0,8}; chunks at klo and klo+16
    const int bklo = hi ? 8 : 0;
    for (int j = 0; j < 2; ++j) {
      int row = wn * 32 + j * 16 + l15;
      b2[j].q[0] = *(const uint4*)&Bs[cur][row * TK + bklo];
      b2[j].q[1] = *(const uint4*)&Bs[cur][row * TK + bklo + 16];
    }
    for (int i = 0; i < 2; ++i)
      for (int j = 0; j < 2; ++j)
        acc[i][j].v = wmma_bf16(a[i].v, b2[j].v, acc[i][j].v);
    __syncthreads();                            // done reading buf before reuse
    cur ^= 1;
  }

  // epilogue: C/D layout => vgpr r -> M = r (lanes 0-15) or r+8 (lanes 16-31)
  for (int i = 0; i < 2; ++i)
    for (int j = 0; j < 2; ++j) {
      int n  = nBase + wn * 32 + j * 16 + l15;
      float bv = bias ? bias[n] : 0.f;
      for (int r = 0; r < 8; ++r) {
        int m = mBase + wm * 32 + i * 16 + (hi ? r + 8 : r);
        float val = acc[i][j].f[r] + bv;
        if (flags & GF_POS)  val += pos[(m & 1023) * Nout + n];
        if (flags & GF_GELU) val = 0.5f * val * (1.f + erff(val * 0.70710678f));
        size_t idx = (size_t)m * Nout + n;
        if (outF) outF[idx] = val;
        if (outB) outB[idx] = f2bf(val);
      }
    }
}

// --------------------- fused residual + LayerNorm --------------------------

__global__ __launch_bounds__(256) void k_ln_residual(
    const float* __restrict__ x, const float* __restrict__ del,
    const float* __restrict__ g, const float* __restrict__ bta,
    float* __restrict__ outF, unsigned short* __restrict__ outB) {
  const int row = blockIdx.x, tid = threadIdx.x;
  const float* xr = x + (size_t)row * 768;
  const float* dr = del + (size_t)row * 768;
  float v[3];
  float s = 0.f;
  for (int i = 0; i < 3; ++i) {
    v[i] = xr[tid + i * 256] + dr[tid + i * 256];
    s += v[i];
  }
  __shared__ float red[256];
  red[tid] = s;
  __syncthreads();
  for (int st = 128; st > 0; st >>= 1) {
    if (tid < st) red[tid] += red[tid + st];
    __syncthreads();
  }
  float mu = red[0] * (1.f / 768.f);
  __syncthreads();
  s = 0.f;
  for (int i = 0; i < 3; ++i) { float d2 = v[i] - mu; s += d2 * d2; }
  red[tid] = s;
  __syncthreads();
  for (int st = 128; st > 0; st >>= 1) {
    if (tid < st) red[tid] += red[tid + st];
    __syncthreads();
  }
  float inv = rsqrtf(red[0] * (1.f / 768.f) + 1e-5f);
  for (int i = 0; i < 3; ++i) {
    int c = tid + i * 256;
    float y = (v[i] - mu) * inv * g[c] + bta[c];
    outF[(size_t)row * 768 + c] = y;
    outB[(size_t)row * 768 + c] = f2bf(y);
  }
}

// --------------------- flash attention (WMMA, online softmax) --------------
// Block = 128 threads (4 waves); each wave: one 16-row Q tile.
// Per 32-key step: 4 score WMMAs + 4 P*V WMMAs.

__global__ __launch_bounds__(128) void k_flash_attn(
    const unsigned short* __restrict__ qkv,   // [B,N,3D] bf16
    const unsigned short* __restrict__ vT,    // [B,H,64,N] bf16
    const float* __restrict__ bNN,            // [N,N] additive bias
    unsigned short* __restrict__ outB) {      // [B,N,D] bf16
  __shared__ __align__(16) float          sS[4][16 * 32];
  __shared__ __align__(16) unsigned short sP[4][16 * 32];
  __shared__ float sF[4][16];
  __shared__ float sL[4][16];

  const int tid = threadIdx.x, lane = tid & 31, wv = tid >> 5;
  const int tile = blockIdx.x * 4 + wv;
  const int qt = tile & 63;                 // N/16 = 64 q tiles
  const int h  = (tile >> 6) % 12;
  const int b  = tile / (64 * 12);
  const int l15 = lane & 15, hi = lane >> 4;
  const int qbase = qt * 16;
  const float scale = 0.125f;               // 1/sqrt(64)

  // hoist Q A-fragments (16x64 => two K-steps of 32)
  Frag16 aQ[2];
  {
    int row = qbase + l15;
    size_t base = ((size_t)(b * 1024 + row)) * 2304 + h * 64;
    int klo = hi ? 8 : 0;
    for (int s2 = 0; s2 < 2; ++s2) {
      aQ[s2].q[0] = *(const uint4*)&qkv[base + s2 * 32 + klo];
      aQ[s2].q[1] = *(const uint4*)&qkv[base + s2 * 32 + klo + 16];
    }
  }

  AccU accO[4];
  for (int t = 0; t < 4; ++t)
    for (int r = 0; r < 8; ++r) accO[t].f[r] = 0.f;
  float m_r = -1e30f, l_r = 0.f;            // stats owned by lanes 0..15

  for (int kb = 0; kb < 1024; kb += 32) {
    // ----- scores: S = Q * K^T (two 16-key tiles) -----
    for (int j = 0; j < 2; ++j) {
      int key = kb + j * 16 + l15;
      size_t kbase = ((size_t)(b * 1024 + key)) * 2304 + 768 + h * 64;
      int klo = hi ? 8 : 0;
      AccU s;
      for (int r = 0; r < 8; ++r) s.f[r] = 0.f;
      for (int s2 = 0; s2 < 2; ++s2) {
        Frag16 bK;
        bK.q[0] = *(const uint4*)&qkv[kbase + s2 * 32 + klo];
        bK.q[1] = *(const uint4*)&qkv[kbase + s2 * 32 + klo + 16];
        s.v = wmma_bf16(aQ[s2].v, bK.v, s.v);
      }
      for (int r = 0; r < 8; ++r) {
        int m = hi ? r + 8 : r;
        int n = j * 16 + l15;
        sS[wv][m * 32 + n] =
            s.f[r] * scale + bNN[(size_t)(qbase + m) * 1024 + kb + n];
      }
    }
    __syncthreads();

    // ----- online softmax update (lane r owns row r) -----
    if (lane < 16) {
      float rm = -1e30f;
      for (int c = 0; c < 32; ++c) rm = fmaxf(rm, sS[wv][lane * 32 + c]);
      float mn  = fmaxf(m_r, rm);
      float fac = __expf(m_r - mn);
      float ls  = 0.f;
      for (int c = 0; c < 32; ++c) {
        float p = __expf(sS[wv][lane * 32 + c] - mn);
        sP[wv][lane * 32 + c] = f2bf(p);
        ls += p;
      }
      l_r = l_r * fac + ls;
      m_r = mn;
      sF[wv][lane] = fac;
    }
    __syncthreads();

    // rescale O accumulators
    for (int r = 0; r < 8; ++r) {
      float f = sF[wv][hi ? r + 8 : r];
      for (int t = 0; t < 4; ++t) accO[t].f[r] *= f;
    }

    // ----- O += P * V -----
    Frag16 aP;
    {
      int klo = hi ? 8 : 0;
      aP.q[0] = *(const uint4*)&sP[wv][l15 * 32 + klo];
      aP.q[1] = *(const uint4*)&sP[wv][l15 * 32 + klo + 16];
    }
    const int vklo = hi ? 8 : 0;
    for (int t = 0; t < 4; ++t) {
      int hd = t * 16 + l15;
      const uint4* vp =
          (const uint4*)&vT[((size_t)((b * 12 + h) * 64 + hd)) * 1024 + kb + vklo];
      Frag16 bV;
      bV.q[0] = vp[0];
      bV.q[1] = vp[2];
      accO[t].v = wmma_bf16(aP.v, bV.v, accO[t].v);
    }
    __syncthreads();
  }

  if (lane < 16) sL[wv][lane] = l_r;
  __syncthreads();

  for (int r = 0; r < 8; ++r) {
    int m = hi ? r + 8 : r;
    float inv = 1.f / sL[wv][m];
    size_t base = ((size_t)(b * 1024 + qbase + m)) * 768 + h * 64;
    for (int t = 0; t < 4; ++t)
      outB[base + t * 16 + l15] = f2bf(accO[t].f[r] * inv);
  }
}

// ------------------------------- launch ------------------------------------

extern "C" void kernel_launch(void* const* d_in, const int* in_sizes, int n_in,
                              void* d_out, int out_size, void* d_ws, size_t ws_size,
                              hipStream_t stream) {
  enum { Bc = 4, Nc = 1024, PDc = 768, Dc = 768, Hc = 12, Lc = 6, Fc = 3072, Vc = 1024 };
  const int BNt = Bc * Nc;  // 4096 tokens

  const float* patches    = (const float*)d_in[0];
  const int*   mask       = (const int*)  d_in[1];
  const float* mask_token = (const float*)d_in[2];
  const float* patch_w    = (const float*)d_in[3];
  const float* patch_b    = (const float*)d_in[4];
  const float* pos_emb    = (const float*)d_in[5];
  const float* rel_bias   = (const float*)d_in[6];
  const float* in_w       = (const float*)d_in[7];
  const float* in_b       = (const float*)d_in[8];
  const float* attn_ow    = (const float*)d_in[9];
  const float* attn_ob    = (const float*)d_in[10];
  const float* ln1_s      = (const float*)d_in[11];
  const float* ln1_b      = (const float*)d_in[12];
  const float* ffn_w1     = (const float*)d_in[13];
  const float* ffn_b1     = (const float*)d_in[14];
  const float* ffn_w2     = (const float*)d_in[15];
  const float* ffn_b2     = (const float*)d_in[16];
  const float* ln2_s      = (const float*)d_in[17];
  const float* ln2_b      = (const float*)d_in[18];
  const float* head_w     = (const float*)d_in[19];
  const float* head_b     = (const float*)d_in[20];
  (void)in_sizes; (void)n_in; (void)out_size; (void)ws_size;

  char* w = (char*)d_ws;
  size_t off = 0;
  auto alloc = [&](size_t bytes) -> char* {
    char* p = w + off;
    off += (bytes + 255) & ~(size_t)255;
    return p;
  };
  unsigned short* pw_bf  = (unsigned short*)alloc((size_t)Dc * PDc * 2);
  unsigned short* inw_bf = (unsigned short*)alloc((size_t)Lc * 3 * Dc * Dc * 2);
  unsigned short* ow_bf  = (unsigned short*)alloc((size_t)Lc * Dc * Dc * 2);
  unsigned short* w1_bf  = (unsigned short*)alloc((size_t)Lc * Fc * Dc * 2);
  unsigned short* w2_bf  = (unsigned short*)alloc((size_t)Lc * Dc * Fc * 2);
  unsigned short* hw_bf  = (unsigned short*)alloc((size_t)Vc * Dc * 2);
  unsigned short* pin    = (unsigned short*)alloc((size_t)BNt * PDc * 2);
  float*          xf     = (float*)         alloc((size_t)BNt * Dc * 4);
  unsigned short* xb     = (unsigned short*)alloc((size_t)BNt * Dc * 2);
  unsigned short* qkvb   = (unsigned short*)alloc((size_t)BNt * 3 * Dc * 2);
  unsigned short* vT     = (unsigned short*)alloc((size_t)Bc * Hc * 64 * Nc * 2);
  unsigned short* attnb  = (unsigned short*)alloc((size_t)BNt * Dc * 2);
  float*          of     = (float*)         alloc((size_t)BNt * Dc * 4);
  unsigned short* hb     = (unsigned short*)alloc((size_t)BNt * Fc * 2);
  float*          ff     = (float*)         alloc((size_t)BNt * Dc * 4);
  float*          bNN    = (float*)         alloc((size_t)Nc * Nc * 4);

  auto cvt = [&](const float* s, unsigned short* d2, size_t n) {
    int g = (int)((n + 2047) / 2048);
    if (g > 4096) g = 4096;
    k_f32_to_bf16<<<g, 256, 0, stream>>>(s, d2, n);
  };
  cvt(patch_w, pw_bf, (size_t)Dc * PDc);
  cvt(in_w,    inw_bf, (size_t)Lc * 3 * Dc * Dc);
  cvt(attn_ow, ow_bf,  (size_t)Lc * Dc * Dc);
  cvt(ffn_w1,  w1_bf,  (size_t)Lc * Fc * Dc);
  cvt(ffn_w2,  w2_bf,  (size_t)Lc * Dc * Fc);
  cvt(head_w,  hw_bf,  (size_t)Vc * Dc);

  k_rel_bias<<<(Nc * Nc) / 256, 256, 0, stream>>>(rel_bias, bNN);
  k_patch_prep<<<4096, 256, 0, stream>>>(patches, mask, mask_token, pin, BNt * PDc);

  // patch embedding: x = pin @ patch_w^T + patch_b + pos_emb
  k_gemm_bf16<<<dim3(Dc / TN, BNt / TM), 256, 0, stream>>>(
      pin, pw_bf, patch_b, xf, xb, pos_emb, BNt, PDc, Dc, GF_POS);

  for (int l = 0; l < Lc; ++l) {
    // qkv (bf16 only)
    k_gemm_bf16<<<dim3(3 * Dc / TN, BNt / TM), 256, 0, stream>>>(
        xb, inw_bf + (size_t)l * 3 * Dc * Dc, in_b + l * 3 * Dc,
        nullptr, qkvb, nullptr, BNt, Dc, 3 * Dc, 0);
    k_build_vt<<<3072, 256, 0, stream>>>(qkvb, vT, Bc * Hc * 64 * Nc);
    k_flash_attn<<<Bc * Hc * (Nc / 16) / 4, 128, 0, stream>>>(qkvb, vT, bNN, attnb);
    // attention output projection (f32)
    k_gemm_bf16<<<dim3(Dc / TN, BNt / TM), 256, 0, stream>>>(
        attnb, ow_bf + (size_t)l * Dc * Dc, attn_ob + l * Dc,
        of, nullptr, nullptr, BNt, Dc, Dc, 0);
    k_ln_residual<<<BNt, 256, 0, stream>>>(xf, of, ln1_s + l * Dc, ln1_b + l * Dc, xf, xb);
    // FFN up + GELU (bf16 only)
    k_gemm_bf16<<<dim3(Fc / TN, BNt / TM), 256, 0, stream>>>(
        xb, w1_bf + (size_t)l * Fc * Dc, ffn_b1 + l * Fc,
        nullptr, hb, nullptr, BNt, Dc, Fc, GF_GELU);
    // FFN down (f32)
    k_gemm_bf16<<<dim3(Dc / TN, BNt / TM), 256, 0, stream>>>(
        hb, w2_bf + (size_t)l * Dc * Fc, ffn_b2 + l * Dc,
        ff, nullptr, nullptr, BNt, Fc, Dc, 0);
    k_ln_residual<<<BNt, 256, 0, stream>>>(xf, ff, ln2_s + l * Dc, ln2_b + l * Dc, xf, xb);
  }

  // head: logits -> d_out (f32)
  k_gemm_bf16<<<dim3(Vc / TN, BNt / TM), 256, 0, stream>>>(
      xb, hw_bf, head_b, (float*)d_out, nullptr, nullptr, BNt, Dc, Vc, 0);
  // mask echo after logits
  k_mask_out<<<(BNt + 255) / 256, 256, 0, stream>>>(mask, (float*)d_out + (size_t)BNt * Vc, BNt);
}
